// ContextAM_71468255805545
// MI455X (gfx1250) — matrix operand order repaired and checked by reading, not compiled, not run on hardware
//
#include <hip/hip_runtime.h>
#include <hip/hip_bf16.h>
#include <cstdint>

typedef __attribute__((ext_vector_type(16))) _Float16 v16h;
typedef __attribute__((ext_vector_type(8)))  _Float16 v8h;
typedef __attribute__((ext_vector_type(8)))  float    v8f;
typedef __attribute__((ext_vector_type(4)))  unsigned int u32x4;
typedef __attribute__((ext_vector_type(8)))  int i32x8;
typedef __attribute__((ext_vector_type(4)))  int i32x4;

#define NB 4
#define NC 64
#define NPIX 9216   // 96*96
#define NC8 8

#if defined(__gfx1250__) && __has_builtin(__builtin_amdgcn_tensor_load_to_lds)
#define USE_TDM 1
#else
#define USE_TDM 0
#endif

// ---- order-preserving float<->uint encoding for atomic max ----
__device__ __forceinline__ unsigned encf(float f) {
    unsigned u = __float_as_uint(f);
    return (u & 0x80000000u) ? ~u : (u | 0x80000000u);
}
__device__ __forceinline__ float decf(unsigned k) {
    return (k & 0x80000000u) ? __uint_as_float(k & 0x7FFFFFFFu)
                             : __uint_as_float(~k);
}
__device__ __forceinline__ float fast_sigmoid(float x) {
    // v_exp_f32 + v_add + v_rcp_f32 (no IEEE div expansion)
    return __builtin_amdgcn_rcpf(1.0f + __expf(-x));
}

#if USE_TDM
// Build a 2-D TDM descriptor (D#) and issue tensor_load_to_lds (6-arg form).
// word0_g1 carries workgroup_mask/data_size/pad fields (bits [31:0] of group1).
__device__ __forceinline__ void tdm_load_2d(
    unsigned lds_addr, unsigned long long gaddr_bytes,
    unsigned tensor_d0, unsigned tensor_d1,
    unsigned tile_d0, unsigned tile_d1,
    unsigned long long stride0_elems, unsigned word0_g1)
{
    u32x4 g0;
    g0[0] = 1u;                                       // count=1, user mode
    g0[1] = lds_addr;                                 // lds_addr [63:32]
    g0[2] = (unsigned)(gaddr_bytes & 0xFFFFFFFFull);  // global_addr[31:0]
    g0[3] = (unsigned)((gaddr_bytes >> 32) & 0x01FFFFFFull) | 0x80000000u; // type=2
    i32x8 g1;
    g1[0] = (int)word0_g1;                                        // mask=0,data_size,pad
    g1[1] = (int)((tensor_d0 & 0xFFFFu) << 16);                   // abar=0 | tdim0.lo
    g1[2] = (int)((tensor_d0 >> 16) | ((tensor_d1 & 0xFFFFu) << 16));
    g1[3] = (int)((tensor_d1 >> 16) | (tile_d0 << 16));           // tdim1.hi | tile_dim0
    g1[4] = (int)tile_d1;                                         // tile_dim1 | tile_dim2=0
    g1[5] = (int)(unsigned)(stride0_elems & 0xFFFFFFFFull);       // dim0_stride lo
    g1[6] = (int)(unsigned)((stride0_elems >> 32) & 0xFFFFull);   // dim0_stride hi | dim1_stride=0
    g1[7] = 0;
    i32x4 z4 = {0, 0, 0, 0};
    i32x8 z8 = {0, 0, 0, 0, 0, 0, 0, 0};
    __builtin_amdgcn_tensor_load_to_lds(g0, g1, z4, z4, z8, 0);
}
#endif

// =====================================================================
// Kernel 0: zero reduction buffers (4*64 sums, 4*64 encoded maxes)
// =====================================================================
__global__ void init_red_kernel(float* __restrict__ sums, unsigned* __restrict__ maxs) {
    int i = threadIdx.x;
    sums[i] = 0.0f;
    maxs[i] = 0u;
}

// =====================================================================
// Kernel 1: QKV 1x1 convs, fp32 accumulate, f16 outputs.
// Q: [b][8][n]   K: transposed [b][n][8] (TDM/coalesced friendly)   V: [b][64][n]
// =====================================================================
__global__ __launch_bounds__(256) void qkv_kernel(
    const float* __restrict__ x,
    const float* __restrict__ wq, const float* __restrict__ bq,
    const float* __restrict__ wk, const float* __restrict__ bk,
    const float* __restrict__ wv, const float* __restrict__ bv,
    _Float16* __restrict__ Qw, _Float16* __restrict__ Kw, _Float16* __restrict__ Vw)
{
    __shared__ float sWQ[8 * 64];
    __shared__ float sWK[8 * 64];
    __shared__ float sWV[64 * 64];
    const int tid = threadIdx.x;
    for (int i = tid; i < 512; i += 256) { sWQ[i] = wq[i]; sWK[i] = wk[i]; }
    for (int i = tid; i < 4096; i += 256) sWV[i] = wv[i];
    __syncthreads();

    const int b = blockIdx.y;
    const int n = blockIdx.x * 256 + tid;

    float qa[8], ka[8], va[64];
    #pragma unroll
    for (int o = 0; o < 8; o++) { qa[o] = bq[o]; ka[o] = bk[o]; }
    #pragma unroll
    for (int o = 0; o < 64; o++) va[o] = bv[o];

    for (int c = 0; c < 64; c++) {
        float xv = x[((size_t)(b * 64 + c)) * NPIX + n];
        #pragma unroll
        for (int o = 0; o < 8; o++) {
            qa[o] = fmaf(sWQ[o * 64 + c], xv, qa[o]);
            ka[o] = fmaf(sWK[o * 64 + c], xv, ka[o]);
        }
        #pragma unroll
        for (int o = 0; o < 64; o++) va[o] = fmaf(sWV[o * 64 + c], xv, va[o]);
    }
    #pragma unroll
    for (int o = 0; o < 8; o++) {
        Qw[(b * 8 + o) * NPIX + n] = (_Float16)qa[o];
        Kw[((size_t)(b * NPIX) + n) * 8 + o] = (_Float16)ka[o];   // transposed, 16B/pixel
    }
    #pragma unroll
    for (int o = 0; o < 64; o++) Vw[(b * 64 + o) * NPIX + n] = (_Float16)va[o];
}

// =====================================================================
// Kernel 2: fused attention, double-buffered K/V tiles (TDM or manual),
// 6 WMMAs per 32-m chunk per wave, att never leaves LDS.
// =====================================================================
__global__ __launch_bounds__(256) void attn_kernel(
    const _Float16* __restrict__ Qg, const _Float16* __restrict__ Kg,
    const _Float16* __restrict__ Vg, const float* __restrict__ x,
    float* __restrict__ out, float* __restrict__ sums, unsigned* __restrict__ maxs)
{
    // KB: [buf][m][c(32, c>=8 pre-zeroed)]  VA: [buf][c][m]
    // ATT: per-wave [n][m]   QA: per-wave [n][c]
    __shared__ __attribute__((aligned(64))) _Float16 KB[2][32][32];
    __shared__ __attribute__((aligned(64))) _Float16 VA[2][64][32];
    __shared__ __attribute__((aligned(64))) _Float16 ATT[8][16][32];
    __shared__ __attribute__((aligned(64))) _Float16 QA[8][16][32];

    const int tid  = threadIdx.x;
    const int w    = tid >> 5;
    const int lane = tid & 31;
    const int lo   = lane & 15;
    const int hi   = lane >> 4;
    const int b    = blockIdx.y;
    const int n0   = (blockIdx.x * 8 + w) * 16;

    // ---- one-time zero fill of KB pad columns (c = 8..31, both buffers) ----
    for (int i = tid; i < 2 * 32 * 24; i += 256) {
        int bufz = i / (32 * 24);
        int r    = (i / 24) % 32;
        int c    = 8 + (i % 24);
        KB[bufz][r][c] = (_Float16)0.0f;
    }

    // ---- stage Q^T A-fragment (per-wave region; wave-internal LDS is in-order) ----
    #pragma unroll
    for (int j = 0; j < 16; j++) {
        int c = hi * 16 + j;
        QA[w][lo][c] = (c < NC8) ? Qg[(b * 8 + c) * NPIX + n0 + lo] : (_Float16)0.0f;
    }
    v16h aq;
    {
        aq = *(const v16h*)&QA[w][lo][hi * 16];
    }

    v8f acc[4];
    #pragma unroll
    for (int ct = 0; ct < 4; ct++) acc[ct] = (v8f)(0.0f);

#if USE_TDM
    const unsigned W0_V = (1u << 16);                                   // data_size=2B
    const unsigned W0_K = (1u << 16) | (1u << 20) | (1u << 22) | (11u << 25);
    // ^ pad_enable, pad_interval code 1 (4 DW = 8 f16 row), pad_amount code 11 (12 DW = 24 f16)
    if (w == 0) {   // buf 0, chunk 0
        tdm_load_2d((unsigned)(size_t)&VA[0][0][0],
                    (unsigned long long)(size_t)(Vg + (size_t)b * 64 * NPIX),
                    NPIX, 64, 32, 64, NPIX, W0_V);
        tdm_load_2d((unsigned)(size_t)&KB[0][0][0],
                    (unsigned long long)(size_t)(Kg + (size_t)b * NPIX * 8),
                    8, NPIX, 8, 32, 8, W0_K);
    }
#else
    const int km2 = tid >> 3;          // 0..31
    const int kc2 = tid & 7;           // 0..7
    const int vc  = tid >> 2;          // 0..63
    const int vm  = (tid & 3) * 8;     // 0,8,16,24
    {   // preload chunk 0 into buf 0
        _Float16 k0 = Kg[((size_t)(b * NPIX) + km2) * 8 + kc2];
        v8h v0 = *(const v8h*)&Vg[(size_t)(b * 64 + vc) * NPIX + vm];
        KB[0][km2][kc2] = k0;
        *(v8h*)&VA[0][vc][vm] = v0;
    }
#endif

    int buf = 0;
    for (int mc = 0; mc < NPIX; mc += 32) {
        const bool have_next = (mc + 32 < NPIX);
#if USE_TDM
        if (w == 0) __builtin_amdgcn_s_wait_tensorcnt(0);   // buf's tiles landed
        __syncthreads();                                    // publish; prior reads of buf^1 done
        if (w == 0 && have_next) {
            tdm_load_2d((unsigned)(size_t)&VA[buf ^ 1][0][0],
                        (unsigned long long)(size_t)(Vg + (size_t)b * 64 * NPIX + (mc + 32)),
                        NPIX, 64, 32, 64, NPIX, W0_V);
            tdm_load_2d((unsigned)(size_t)&KB[buf ^ 1][0][0],
                        (unsigned long long)(size_t)(Kg + ((size_t)b * NPIX + (mc + 32)) * 8),
                        8, NPIX, 8, 32, 8, W0_K);
        }
#else
        __syncthreads();                                    // buf visible; reads of buf^1 done
        _Float16 knext = (_Float16)0.0f;
        v8h vnext;
        if (have_next) {                                    // issued early, waited before stores
            knext = Kg[((size_t)(b * NPIX) + mc + 32 + km2) * 8 + kc2];
            vnext = *(const v8h*)&Vg[(size_t)(b * 64 + vc) * NPIX + mc + 32 + vm];
        }
#endif

        // ---- energy = Q^T K, sigmoid, stage att (per-wave, in-order LDS) ----
        #pragma unroll
        for (int t2 = 0; t2 < 2; t2++) {
            v16h kb = *(const v16h*)&KB[buf][t2 * 16 + lo][hi * 16];
            v8f zero = (v8f)(0.0f);
            v8f e = __builtin_amdgcn_wmma_f32_16x16x32_f16(
                false, aq, false, kb, (short)0, zero, false, false);
            #pragma unroll
            for (int v = 0; v < 8; v++) {
                ATT[w][v + hi * 8][t2 * 16 + lo] = (_Float16)fast_sigmoid(e[v]);
            }
        }
        v16h attf = *(const v16h*)&ATT[w][lo][hi * 16];

        // ---- out[c,n] += V[c,m] * att[n,m], 4 c-tiles ----
        #pragma unroll
        for (int ct = 0; ct < 4; ct++) {
            v16h av = *(const v16h*)&VA[buf][ct * 16 + lo][hi * 16];
            acc[ct] = __builtin_amdgcn_wmma_f32_16x16x32_f16(
                false, av, false, attf, (short)0, acc[ct], false, false);
        }

#if !USE_TDM
        if (have_next) {    // nobody reads buf^1 until next barrier
            KB[buf ^ 1][km2][kc2] = knext;
            *(v8h*)&VA[buf ^ 1][vc][vm] = vnext;
        }
#endif
        buf ^= 1;
    }

    // ---- epilogue: residual, store, CBAM avg/max partials ----
    const int n = n0 + lo;
    #pragma unroll
    for (int ct = 0; ct < 4; ct++) {
        #pragma unroll
        for (int v = 0; v < 8; v++) {
            int c = ct * 16 + v + hi * 8;
            size_t idx = ((size_t)(b * 64 + c)) * NPIX + n;
            float o = acc[ct][v] + x[idx];
            out[idx] = o;
            float ssum = o, smax = o;
            #pragma unroll
            for (int off = 1; off < 16; off <<= 1) {
                ssum += __shfl_xor(ssum, off, 16);
                smax = fmaxf(smax, __shfl_xor(smax, off, 16));
            }
            if (lo == 0) {
                atomicAdd(&sums[b * 64 + c], ssum);
                atomicMax(&maxs[b * 64 + c], encf(smax));
            }
        }
    }
}

// =====================================================================
// Kernel 3: CBAM shared-MLP -> sigmoid channel scale
// =====================================================================
__global__ __launch_bounds__(64) void cbam_scale_kernel(
    const float* __restrict__ sums, const unsigned* __restrict__ maxs,
    const float* __restrict__ w1, const float* __restrict__ w2,
    float* __restrict__ scale)
{
    __shared__ float favg[64], fmax[64], hsum[4];
    const int b = blockIdx.x, c = threadIdx.x;
    favg[c] = sums[b * 64 + c] * (1.0f / (float)NPIX);
    fmax[c] = decf(maxs[b * 64 + c]);
    __syncthreads();
    if (c < 4) {
        float ha = 0.f, hm = 0.f;
        for (int j = 0; j < 64; j++) {
            ha = fmaf(w1[c * 64 + j], favg[j], ha);
            hm = fmaf(w1[c * 64 + j], fmax[j], hm);
        }
        hsum[c] = fmaxf(ha, 0.f) + fmaxf(hm, 0.f);
    }
    __syncthreads();
    float a = 0.f;
    for (int r = 0; r < 4; r++) a = fmaf(w2[c * 4 + r], hsum[r], a);
    scale[b * 64 + c] = fast_sigmoid(a);
}

// =====================================================================
// Kernel 4: apply per-(b,c) channel scale
// =====================================================================
__global__ __launch_bounds__(256) void apply_scale_kernel(
    float* __restrict__ out, const float* __restrict__ scale)
{
    size_t idx = (size_t)blockIdx.x * 256 + threadIdx.x;
    int bc = (int)(idx / NPIX);
    out[idx] *= scale[bc];
}

// =====================================================================
extern "C" void kernel_launch(void* const* d_in, const int* in_sizes, int n_in,
                              void* d_out, int out_size, void* d_ws, size_t ws_size,
                              hipStream_t stream)
{
    const float* x  = (const float*)d_in[0];
    const float* wq = (const float*)d_in[1];
    const float* bq = (const float*)d_in[2];
    const float* wk = (const float*)d_in[3];
    const float* bk = (const float*)d_in[4];
    const float* wv = (const float*)d_in[5];
    const float* bv = (const float*)d_in[6];
    const float* w1 = (const float*)d_in[7];
    const float* w2 = (const float*)d_in[8];
    float* out = (float*)d_out;

    const size_t QK_BYTES = (size_t)NB * 8 * NPIX * 2;    // 589824
    const size_t V_BYTES  = (size_t)NB * 64 * NPIX * 2;   // 4718592
    char* ws = (char*)d_ws;
    _Float16* Qw   = (_Float16*)(ws);
    _Float16* Kw   = (_Float16*)(ws + QK_BYTES);
    _Float16* Vw   = (_Float16*)(ws + 2 * QK_BYTES);
    float*    sums = (float*)   (ws + 2 * QK_BYTES + V_BYTES);
    unsigned* maxs = (unsigned*)(ws + 2 * QK_BYTES + V_BYTES + 1024);
    float*    scl  = (float*)   (ws + 2 * QK_BYTES + V_BYTES + 2048);

    init_red_kernel<<<1, 256, 0, stream>>>(sums, maxs);
    qkv_kernel<<<dim3(NPIX / 256, NB), 256, 0, stream>>>(
        x, wq, bq, wk, bk, wv, bv, Qw, Kw, Vw);
    attn_kernel<<<dim3(NPIX / 128, NB), 256, 0, stream>>>(
        Qw, Kw, Vw, x, out, sums, maxs);
    cbam_scale_kernel<<<dim3(NB), 64, 0, stream>>>(sums, maxs, w1, w2, scl);
    apply_scale_kernel<<<dim3((NB * NC * NPIX) / 256), 256, 0, stream>>>(out, scl);
}
